// RelativeMultiHeadAttention_5325759447455
// MI455X (gfx1250) — matrix operand order repaired and checked by reading, not compiled, not run on hardware
//
#include <hip/hip_runtime.h>
#include <cmath>

#define DM 512
#define HD 8
#define DHD 64
#define SS 1024
#define BBATCH 8
#define EPSF 6.1e-05f
#define INV_SQRT_D 0.0441941738241592f  // 1/sqrt(512)

typedef __attribute__((ext_vector_type(16))) __bf16 v16bf;
typedef __attribute__((ext_vector_type(8)))  __bf16 v8bf;
typedef __attribute__((ext_vector_type(8)))  float  v8f;

// ---------------- WMMA fragment helpers (CDNA5 16x16x32 bf16) ----------------
// A (16x32, MxK): lane m = lane&15, khalf = lane>>4; element i holds K = 16*kh + i
// -> two 16-byte loads per lane.
__device__ __forceinline__ v16bf load_a_frag(const __bf16* p, int kh) {
  v8bf lo = *(const v8bf*)(p + 16 * kh);
  v8bf hi = *(const v8bf*)(p + 16 * kh + 8);
  v16bf a;
#pragma unroll
  for (int i = 0; i < 8; i++) { a[i] = lo[i]; a[8 + i] = hi[i]; }
  return a;
}
// B (32x16, KxN) stored as N x K row-major: lane n = lane&15, khalf = lane>>4;
// VGPR i holds K = 16*kh + i.
__device__ __forceinline__ v16bf load_b_frag(const __bf16* p, int kh) {
  v8bf lo = *(const v8bf*)(p + 16 * kh);
  v8bf hi = *(const v8bf*)(p + 16 * kh + 8);
  v16bf b;
#pragma unroll
  for (int i = 0; i < 8; i++) { b[i] = lo[i]; b[8 + i] = hi[i]; }
  return b;
}
__device__ __forceinline__ v8f wmma_bf16(v16bf a, v16bf b, v8f c) {
  return __builtin_amdgcn_wmma_f32_16x16x32_bf16(false, a, false, b, (short)0, c,
                                                 false, false);
}
__device__ __forceinline__ float wred_sum(float v) {
#pragma unroll
  for (int o = 16; o; o >>= 1) v += __shfl_xor(v, o, 32);
  return v;
}
__device__ __forceinline__ float wred_max(float v) {
#pragma unroll
  for (int o = 16; o; o >>= 1) v = fmaxf(v, __shfl_xor(v, o, 32));
  return v;
}

// ---------------- LayerNorm -> bf16, one wave per row ----------------
__global__ void k_layernorm(const float* __restrict__ x,
                            const float* __restrict__ gamma,
                            const float* __restrict__ beta,
                            __bf16* __restrict__ xn) {
  const int wid = threadIdx.x >> 5, lane = threadIdx.x & 31;
  const size_t row = (size_t)blockIdx.x * 8 + wid;  // < B*S = 8192
  const float* xr = x + row * DM;
  float r[16];
  float s = 0.f;
#pragma unroll
  for (int i = 0; i < 16; i++) { r[i] = xr[lane + 32 * i]; s += r[i]; }
  const float mean = wred_sum(s) * (1.0f / DM);
  float vs = 0.f;
#pragma unroll
  for (int i = 0; i < 16; i++) { float d = r[i] - mean; vs += d * d; }
  const float rstd = rsqrtf(wred_sum(vs) * (1.0f / DM) + EPSF);
  __bf16* o = xn + row * DM;
#pragma unroll
  for (int i = 0; i < 16; i++) {
    int c = lane + 32 * i;
    o[c] = (__bf16)((r[i] - mean) * rstd * gamma[c] + beta[c]);
  }
}

// ---------------- f32 -> bf16 convert ----------------
__global__ void k_cvt(const float* __restrict__ in, __bf16* __restrict__ out, int n) {
  int i = blockIdx.x * blockDim.x + threadIdx.x;
  if (i < n) out[i] = (__bf16)in[i];
}

// ---------------- sinusoidal positional encoding -> bf16 ----------------
__global__ void k_posenc(__bf16* __restrict__ pe) {
  const int s = blockIdx.x;        // 0..S-1
  const int i = threadIdx.x;       // 0..255 (= D/2)
  // inv_freq = 10000^{-2i/D} = exp(-(2i) * ln(10000)/D)
  const float inv = __expf(-(float)(2 * i) * (9.210340371976184f / (float)DM));
  const float ang = (float)s * inv;
  pe[(size_t)s * DM + 2 * i]     = (__bf16)sinf(ang);
  pe[(size_t)s * DM + 2 * i + 1] = (__bf16)cosf(ang);
}

// ---------------- generic WMMA linear: Y = A @ W^T + bias ----------------
// block = 128 (4 waves); each wave does a 16x64 output tile (4 accumulators,
// A fragment reused 4x). Software-pipelined: iteration i's WMMAs consume
// fragments loaded at iteration i-1, so current-iteration loads overlap the
// matrix math (fragment-in-flight and fragment-in-use are simultaneously
// live -> RA cannot coalesce them, waits tolerate in-flight loads).
// mode: 0 = write (y+u), (y+v) bf16; 1 = bf16; 2 = bf16 transposed V store
// [b][h][dh][t]; 3 = f32 out.
__global__ void k_linear(const __bf16* __restrict__ A,
                         const __bf16* __restrict__ Wt,
                         const float* __restrict__ bias, int mode,
                         __bf16* __restrict__ out0, __bf16* __restrict__ out1,
                         const float* __restrict__ ub,
                         const float* __restrict__ vb,
                         float* __restrict__ outf) {
  const int lane = threadIdx.x & 31;
  const int w = threadIdx.x >> 5;
  const int row = lane & 15;
  const int kh = lane >> 4;
  const int m0 = blockIdx.x * 16;
  const int n0 = blockIdx.y * 256 + w * 64;
  v8f acc0 = {}, acc1 = {}, acc2 = {}, acc3 = {};
  const __bf16* arow = A + (size_t)(m0 + row) * DM;
  const __bf16* wrow0 = Wt + (size_t)(n0 + row) * DM;
  const __bf16* wrow1 = wrow0 + (size_t)16 * DM;
  const __bf16* wrow2 = wrow0 + (size_t)32 * DM;
  const __bf16* wrow3 = wrow0 + (size_t)48 * DM;
  // prologue
  v16bf a  = load_a_frag(arow, kh);
  v16bf b0 = load_b_frag(wrow0, kh);
  v16bf b1 = load_b_frag(wrow1, kh);
  v16bf b2 = load_b_frag(wrow2, kh);
  v16bf b3 = load_b_frag(wrow3, kh);
  for (int k0 = 32; k0 < DM; k0 += 32) {
    v16bf an  = load_a_frag(arow + k0, kh);
    v16bf bn0 = load_b_frag(wrow0 + k0, kh);
    v16bf bn1 = load_b_frag(wrow1 + k0, kh);
    v16bf bn2 = load_b_frag(wrow2 + k0, kh);
    v16bf bn3 = load_b_frag(wrow3 + k0, kh);
    acc0 = wmma_bf16(a, b0, acc0);
    acc1 = wmma_bf16(a, b1, acc1);
    acc2 = wmma_bf16(a, b2, acc2);
    acc3 = wmma_bf16(a, b3, acc3);
    a = an; b0 = bn0; b1 = bn1; b2 = bn2; b3 = bn3;
  }
  acc0 = wmma_bf16(a, b0, acc0);
  acc1 = wmma_bf16(a, b1, acc1);
  acc2 = wmma_bf16(a, b2, acc2);
  acc3 = wmma_bf16(a, b3, acc3);
  v8f accs[4] = {acc0, acc1, acc2, acc3};
#pragma unroll
  for (int t = 0; t < 4; t++) {
#pragma unroll
    for (int r = 0; r < 8; r++) {
      const int m = m0 + r + 8 * kh;
      const int n = n0 + 16 * t + (lane & 15);
      float val = accs[t][r] + (bias ? bias[n] : 0.0f);
      if (mode == 0) {
        out0[(size_t)m * DM + n] = (__bf16)(val + ub[n]);
        out1[(size_t)m * DM + n] = (__bf16)(val + vb[n]);
      } else if (mode == 1) {
        out0[(size_t)m * DM + n] = (__bf16)val;
      } else if (mode == 2) {
        const int b = m >> 10, tt = m & (SS - 1);
        const int h = n >> 6, dh = n & 63;
        out0[(((size_t)b * HD + h) * DHD + dh) * SS + tt] = (__bf16)val;
      } else {
        outf[(size_t)m * DM + n] = val;
      }
    }
  }
}

// ---------------- AC scores: scores[h][s][t] = (q+u)[s] . k[t] ----------------
// block = 128 (4 waves); wave w handles t-tile by*64 + w*16. Per batch b.
// All fragments (K=64 -> 2 k-steps) loaded up front, then 2 WMMAs.
__global__ void k_score_ac(const __bf16* __restrict__ qu,
                           const __bf16* __restrict__ kk,
                           float* __restrict__ scores, int b) {
  const int lane = threadIdx.x & 31, w = threadIdx.x >> 5;
  const int row = lane & 15, kh = lane >> 4;
  const int s0 = blockIdx.x * 16;
  const int t0 = blockIdx.y * 64 + w * 16;
  const int h = blockIdx.z;
  const __bf16* arow = qu + (size_t)(b * SS + s0 + row) * DM + h * DHD;
  const __bf16* brow = kk + (size_t)(b * SS + t0 + row) * DM + h * DHD;
  v16bf a0 = load_a_frag(arow, kh);
  v16bf a1 = load_a_frag(arow + 32, kh);
  v16bf b0 = load_b_frag(brow, kh);
  v16bf b1 = load_b_frag(brow + 32, kh);
  v8f acc = {};
  acc = wmma_bf16(a0, b0, acc);
  acc = wmma_bf16(a1, b1, acc);
#pragma unroll
  for (int r = 0; r < 8; r++)
    scores[((size_t)h * SS + s0 + r + 8 * kh) * SS + t0 + (lane & 15)] = acc[r];
}

// ---------------- BD scores with exact rel_shift scatter-add ----------------
// R[s,j] = (q+v)[s] . p[j].  rel_shift mapping (exactly one destination each):
//   j >= S-1-s           -> scores[s][j+s-(S-1)] += R[s,j]
//   j <= S-2-s && s >= 1 -> scores[s-1][j+s+1]   += R[s,j]
//   (dest t == s+1 gets the implicit zero pad: no contribution)
__global__ void k_score_bd(const __bf16* __restrict__ qv,
                           const __bf16* __restrict__ pp,
                           float* __restrict__ scores, int bIdx) {
  const int lane = threadIdx.x & 31, w = threadIdx.x >> 5;
  const int row = lane & 15, kh = lane >> 4;
  const int s0 = blockIdx.x * 16;
  const int j0 = blockIdx.y * 64 + w * 16;
  const int h = blockIdx.z;
  const __bf16* arow = qv + (size_t)(bIdx * SS + s0 + row) * DM + h * DHD;
  const __bf16* brow = pp + (size_t)(j0 + row) * DM + h * DHD;
  v16bf a0 = load_a_frag(arow, kh);
  v16bf a1 = load_a_frag(arow + 32, kh);
  v16bf b0 = load_b_frag(brow, kh);
  v16bf b1 = load_b_frag(brow + 32, kh);
  v8f acc = {};
  acc = wmma_bf16(a0, b0, acc);
  acc = wmma_bf16(a1, b1, acc);
#pragma unroll
  for (int r = 0; r < 8; r++) {
    const int s = s0 + r + 8 * kh;
    const int j = j0 + (lane & 15);
    const float val = acc[r];
    if (j >= (SS - 1) - s) {
      const int t = j + s - (SS - 1);
      scores[((size_t)h * SS + s) * SS + t] += val;
    } else if (s >= 1) {
      const int t = j + s + 1;
      scores[((size_t)h * SS + s - 1) * SS + t] += val;
    }
  }
}

// ---------------- softmax row, write bf16 probs in place ----------------
__global__ void k_softmax(float* __restrict__ scores,
                          const unsigned char* __restrict__ mask, int b) {
  __shared__ float sm[8];
  const int rowid = blockIdx.x;              // h*S + s
  const int s = rowid & (SS - 1);
  float* base = scores + (size_t)rowid * SS;
  const unsigned char* mrow = mask + ((size_t)b * SS + s) * SS;
  const int t0 = threadIdx.x * 4;
  const int lane = threadIdx.x & 31, wid = threadIdx.x >> 5;
  float v[4];
#pragma unroll
  for (int i = 0; i < 4; i++) {
    float xv = base[t0 + i] * INV_SQRT_D;
    if (mrow[t0 + i]) xv = -1e30f;
    v[i] = xv;
  }
  float m = wred_max(fmaxf(fmaxf(v[0], v[1]), fmaxf(v[2], v[3])));
  if (lane == 0) sm[wid] = m;
  __syncthreads();
  float rowmax = sm[0];
#pragma unroll
  for (int i = 1; i < 8; i++) rowmax = fmaxf(rowmax, sm[i]);
  __syncthreads();
  float sum = 0.f;
#pragma unroll
  for (int i = 0; i < 4; i++) { v[i] = __expf(v[i] - rowmax); sum += v[i]; }
  sum = wred_sum(sum);
  if (lane == 0) sm[wid] = sum;
  __syncthreads();
  float total = 0.f;
#pragma unroll
  for (int i = 0; i < 8; i++) total += sm[i];
  const float inv = 1.0f / total;
  __syncthreads();  // all f32 reads done before bf16 overwrite
  __bf16* ob = (__bf16*)base;
#pragma unroll
  for (int i = 0; i < 4; i++) ob[t0 + i] = (__bf16)(v[i] * inv);
}

// ---------------- ctx[b,s,h,:] = probs[h,s,:] @ v ----------------
// probs live as bf16 in the first half of each f32 score row (stride 2*S bf16).
// block = 128 (4 waves); wave w handles head h = by*4 + w (its full 64 cols).
// Software-pipelined like k_linear.
__global__ void k_attn_v(const float* __restrict__ scores,
                         const __bf16* __restrict__ vT,
                         __bf16* __restrict__ ctx, int b) {
  const int lane = threadIdx.x & 31, w = threadIdx.x >> 5;
  const int row = lane & 15, kh = lane >> 4;
  const int s0 = blockIdx.x * 16;
  const int h = blockIdx.y * 4 + w;
  const __bf16* pb = (const __bf16*)scores;
  v8f acc0 = {}, acc1 = {}, acc2 = {}, acc3 = {};
  const __bf16* arow = pb + ((size_t)h * SS + s0 + row) * (2 * SS);
  const __bf16* vrow0 = vT + (((size_t)b * HD + h) * DHD + row) * SS;
  const __bf16* vrow1 = vrow0 + (size_t)16 * SS;
  const __bf16* vrow2 = vrow0 + (size_t)32 * SS;
  const __bf16* vrow3 = vrow0 + (size_t)48 * SS;
  // prologue
  v16bf a  = load_a_frag(arow, kh);
  v16bf b0 = load_b_frag(vrow0, kh);
  v16bf b1 = load_b_frag(vrow1, kh);
  v16bf b2 = load_b_frag(vrow2, kh);
  v16bf b3 = load_b_frag(vrow3, kh);
  for (int k0 = 32; k0 < SS; k0 += 32) {
    v16bf an  = load_a_frag(arow + k0, kh);
    v16bf bn0 = load_b_frag(vrow0 + k0, kh);
    v16bf bn1 = load_b_frag(vrow1 + k0, kh);
    v16bf bn2 = load_b_frag(vrow2 + k0, kh);
    v16bf bn3 = load_b_frag(vrow3 + k0, kh);
    acc0 = wmma_bf16(a, b0, acc0);
    acc1 = wmma_bf16(a, b1, acc1);
    acc2 = wmma_bf16(a, b2, acc2);
    acc3 = wmma_bf16(a, b3, acc3);
    a = an; b0 = bn0; b1 = bn1; b2 = bn2; b3 = bn3;
  }
  acc0 = wmma_bf16(a, b0, acc0);
  acc1 = wmma_bf16(a, b1, acc1);
  acc2 = wmma_bf16(a, b2, acc2);
  acc3 = wmma_bf16(a, b3, acc3);
  v8f accs[4] = {acc0, acc1, acc2, acc3};
#pragma unroll
  for (int t = 0; t < 4; t++)
#pragma unroll
    for (int r = 0; r < 8; r++)
      ctx[(size_t)(b * SS + s0 + r + 8 * kh) * DM + h * DHD + 16 * t +
          (lane & 15)] = (__bf16)accs[t][r];
}

// ---------------- host launcher ----------------
extern "C" void kernel_launch(void* const* d_in, const int* in_sizes, int n_in,
                              void* d_out, int out_size, void* d_ws,
                              size_t ws_size, hipStream_t stream) {
  const float* x = (const float*)d_in[0];
  const unsigned char* mask = (const unsigned char*)d_in[1];
  const float* Wq = (const float*)d_in[2];
  const float* bq = (const float*)d_in[3];
  const float* Wk = (const float*)d_in[4];
  const float* bk = (const float*)d_in[5];
  const float* Wv = (const float*)d_in[6];
  const float* bv = (const float*)d_in[7];
  const float* Wpos = (const float*)d_in[8];
  const float* Wout = (const float*)d_in[9];
  const float* bout = (const float*)d_in[10];
  const float* ub = (const float*)d_in[11];
  const float* vb = (const float*)d_in[12];
  const float* gamma = (const float*)d_in[13];
  const float* beta = (const float*)d_in[14];
  float* out = (float*)d_out;

  char* base = (char*)d_ws;
  size_t off = 0;
  auto alloc = [&](size_t elems, size_t esz) -> void* {
    void* p = base + off;
    off += ((elems * esz) + 255) & ~(size_t)255;
    return p;
  };
  const size_t NROWS = (size_t)BBATCH * SS;          // 8192
  __bf16* xn   = (__bf16*)alloc(NROWS * DM, 2);      // 8 MB
  __bf16* wqb  = (__bf16*)alloc(DM * DM, 2);
  __bf16* wkb  = (__bf16*)alloc(DM * DM, 2);
  __bf16* wvb  = (__bf16*)alloc(DM * DM, 2);
  __bf16* wpb  = (__bf16*)alloc(DM * DM, 2);
  __bf16* wob  = (__bf16*)alloc(DM * DM, 2);
  __bf16* pe   = (__bf16*)alloc((size_t)SS * DM, 2); // 1 MB
  __bf16* pp   = (__bf16*)alloc((size_t)SS * DM, 2);
  __bf16* qu   = (__bf16*)alloc(NROWS * DM, 2);
  __bf16* qv   = (__bf16*)alloc(NROWS * DM, 2);
  __bf16* kb   = (__bf16*)alloc(NROWS * DM, 2);
  __bf16* vT   = (__bf16*)alloc(NROWS * DM, 2);
  __bf16* ctx  = (__bf16*)alloc(NROWS * DM, 2);
  float*  sc   = (float*)alloc((size_t)HD * SS * SS, 4);  // 32 MB, L2-resident

  k_layernorm<<<1024, 256, 0, stream>>>(x, gamma, beta, xn);
  k_cvt<<<1024, 256, 0, stream>>>(Wq, wqb, DM * DM);
  k_cvt<<<1024, 256, 0, stream>>>(Wk, wkb, DM * DM);
  k_cvt<<<1024, 256, 0, stream>>>(Wv, wvb, DM * DM);
  k_cvt<<<1024, 256, 0, stream>>>(Wpos, wpb, DM * DM);
  k_cvt<<<1024, 256, 0, stream>>>(Wout, wob, DM * DM);
  k_posenc<<<SS, 256, 0, stream>>>(pe);

  dim3 gBig(NROWS / 16, 2);  // (512, 2), 128-thread blocks, 4 waves x 64 cols
  k_linear<<<gBig, 128, 0, stream>>>(xn, wqb, bq, 0, qu, qv, ub, vb, nullptr);
  k_linear<<<gBig, 128, 0, stream>>>(xn, wkb, bk, 1, kb, nullptr, nullptr, nullptr, nullptr);
  k_linear<<<gBig, 128, 0, stream>>>(xn, wvb, bv, 2, vT, nullptr, nullptr, nullptr, nullptr);
  k_linear<<<dim3(SS / 16, 2), 128, 0, stream>>>(pe, wpb, nullptr, 1, pp, nullptr,
                                                 nullptr, nullptr, nullptr);

  for (int b = 0; b < BBATCH; b++) {
    k_score_ac<<<dim3(SS / 16, SS / 64, HD), 128, 0, stream>>>(qu, kb, sc, b);
    k_score_bd<<<dim3(SS / 16, SS / 64, HD), 128, 0, stream>>>(qv, pp, sc, b);
    k_softmax<<<HD * SS, 256, 0, stream>>>(sc, mask, b);
    k_attn_v<<<dim3(SS / 16, 2), 128, 0, stream>>>(sc, vT, ctx, b);
  }

  k_linear<<<gBig, 128, 0, stream>>>(ctx, wob, bout, 3, nullptr, nullptr, nullptr,
                                     nullptr, out);
  (void)in_sizes; (void)n_in; (void)out_size; (void)ws_size;
}
